// T5Attention_81544249082004
// MI455X (gfx1250) — matrix operand order repaired and checked by reading, not compiled, not run on hardware
//
#include <hip/hip_runtime.h>

// ---------------------------------------------------------------------------
// T5 attention for gfx1250 (MI455X), all GEMMs via v_wmma_f32_16x16x32_bf16.
// Weight tiles staged into LDS with global_load_async_to_lds_b128 (ASYNCcnt).
// B=2, S=2048, HID=1024, H=16, DK=64.
// ---------------------------------------------------------------------------

static constexpr int kB   = 2;
static constexpr int kS   = 2048;
static constexpr int kHID = 1024;
static constexpr int kH   = 16;
static constexpr int kDK  = 64;
static constexpr int kM   = kB * kS;   // 4096 rows of x

typedef __attribute__((ext_vector_type(16))) __bf16 bf16x16;
typedef __attribute__((ext_vector_type(8)))  float  f32x8;

union Frag {
  bf16x16 v;
  uint4   q[2];          // 32 bytes = 16 bf16
  unsigned short u[16];
};

__device__ __forceinline__ unsigned short f2bf(float f) {
  unsigned int u = __float_as_uint(f);
  unsigned int r = u + 0x7FFFu + ((u >> 16) & 1u);   // round-to-nearest-even
  return (unsigned short)(r >> 16);
}

__device__ __forceinline__ f32x8 wmma_bf16(Frag a, Frag b, f32x8 c) {
  // D = A(16x32 bf16) * B(32x16 bf16) + C(16x16 f32)
  return __builtin_amdgcn_wmma_f32_16x16x32_bf16(
      /*neg_a=*/false, a.v, /*neg_b=*/false, b.v,
      /*c_mod=*/(short)0, c, /*reuse_a=*/false, /*reuse_b=*/false);
}

// ---------------------------------------------------------------------------
// f32 -> bf16 cast (grid-stride)
// ---------------------------------------------------------------------------
__global__ void cast_bf16_kernel(const float* __restrict__ src,
                                 unsigned short* __restrict__ dst, int n) {
  int i = blockIdx.x * blockDim.x + threadIdx.x;
  int stride = gridDim.x * blockDim.x;
  for (; i < n; i += stride) dst[i] = f2bf(src[i]);
}

// ---------------------------------------------------------------------------
// Q/K/V projection: y = x @ W.T  (M=4096, N=1024, K=1024)
// blockIdx.z selects {Wq,Wk,Wv}. Q,K stored [B][H][S][DK]; V stored
// transposed [B][H][DK][S] so attention's PV B-fragments are contiguous.
// Block: 256 thr (8 waves), tile 128(M) x 64(N); wave tile 32x32.
// Weight tile (64 rows x 32 k, 4KB bf16) staged in LDS via async loads.
// ---------------------------------------------------------------------------
__global__ __launch_bounds__(256)
void proj_qkv_kernel(const unsigned short* __restrict__ xbf,
                     const unsigned short* __restrict__ wq,
                     const unsigned short* __restrict__ wk,
                     const unsigned short* __restrict__ wv,
                     unsigned short* __restrict__ qout,
                     unsigned short* __restrict__ kout,
                     unsigned short* __restrict__ vtout) {
  __shared__ __align__(16) unsigned short tileB[64][32];   // [n][k] bf16, 4KB

  const int which = blockIdx.z;
  const unsigned short* __restrict__ w = (which == 0) ? wq : (which == 1) ? wk : wv;

  const int tid = threadIdx.x, wave = tid >> 5, lane = tid & 31;
  const int lhalf = lane >> 4, l16 = lane & 15;
  const int m0 = blockIdx.x * 128 + (wave & 3) * 32;
  const int nblk = blockIdx.y * 64;
  const int wn = wave >> 2;                 // 0..1, 32-col slice of the tile

  f32x8 acc[2][2];
#pragma unroll
  for (int mt = 0; mt < 2; ++mt)
#pragma unroll
    for (int nt = 0; nt < 2; ++nt)
#pragma unroll
      for (int r = 0; r < 8; ++r) acc[mt][nt][r] = 0.0f;

  // async-copy assignment: 256 threads x 16B = 64x32 bf16 tile
  const unsigned lds_base = (unsigned)(unsigned long long)(void*)&tileB[0][0];
  const int tn = tid >> 2;                  // tile row 0..63
  const int tc = tid & 3;                   // 16B chunk within 64B row
  const unsigned short* gsrc = w + (size_t)(nblk + tn) * kHID + tc * 8;
  const unsigned lds_dst = lds_base + tn * 64 + tc * 16;

  for (int k0 = 0; k0 < kHID; k0 += 32) {
    // stage weight tile into LDS on the async engine (ASYNCcnt-tracked)
    asm volatile("global_load_async_to_lds_b128 %0, %1, off"
                 :: "v"(lds_dst), "v"(gsrc + k0) : "memory");
    asm volatile("s_wait_asynccnt 0" ::: "memory");
    __syncthreads();

    Frag a[2], bb[2];
#pragma unroll
    for (int mt = 0; mt < 2; ++mt) {
      const unsigned short* arow = xbf + (m0 + mt * 16 + l16) * kHID + k0;
      a[mt].q[0] = *(const uint4*)(arow + lhalf * 16);       // K = half*16 .. +7
      a[mt].q[1] = *(const uint4*)(arow + lhalf * 16 + 8);   // K = half*16+8 .. +15
    }
#pragma unroll
    for (int nt = 0; nt < 2; ++nt) {
      const unsigned short* brow = &tileB[wn * 32 + nt * 16 + l16][lhalf * 16];
      bb[nt].q[0] = *(const uint4*)(brow);
      bb[nt].q[1] = *(const uint4*)(brow + 8);
    }
#pragma unroll
    for (int mt = 0; mt < 2; ++mt)
#pragma unroll
      for (int nt = 0; nt < 2; ++nt)
        acc[mt][nt] = wmma_bf16(a[mt], bb[nt], acc[mt][nt]);

    __syncthreads();   // tile consumed before next iteration overwrites it
  }

#pragma unroll
  for (int mt = 0; mt < 2; ++mt)
#pragma unroll
    for (int nt = 0; nt < 2; ++nt)
#pragma unroll
      for (int r = 0; r < 8; ++r) {
        int m = m0 + mt * 16 + lhalf * 8 + r;   // C layout: row=8*(lane/16)+r
        int n = nblk + wn * 32 + nt * 16 + l16; //           col=lane%16
        int b = m >> 11, s = m & (kS - 1);
        int h = n >> 6,  d = n & (kDK - 1);
        unsigned short val = f2bf(acc[mt][nt][r]);
        if (which == 0)      qout[(((b * kH + h) * kS) + s) * kDK + d] = val;
        else if (which == 1) kout[(((b * kH + h) * kS) + s) * kDK + d] = val;
        else                 vtout[(((b * kH + h) * kDK) + d) * kS + s] = val;
      }
}

// ---------------------------------------------------------------------------
// Flash attention per (b,h). Wave = 16 q-rows; kv blocks of 32, causal bound.
// scores = (Q K^T)/8 + rel_bias; online softmax; O += P V  (V transposed).
// ctx written bf16 into [B][S][HID].
// ---------------------------------------------------------------------------
__global__ __launch_bounds__(256)
void attn_kernel(const unsigned short* __restrict__ qbf,   // [B][H][S][DK]
                 const unsigned short* __restrict__ kbf,   // [B][H][S][DK]
                 const unsigned short* __restrict__ vtbf,  // [B][H][DK][S]
                 const int*            __restrict__ mask,  // [B][S]
                 const float*          __restrict__ rel,   // [32][H]
                 unsigned short*       __restrict__ ctx) { // [B][S][HID]
  __shared__ __align__(16) float sh_bias[32];
  __shared__ __align__(16) unsigned short pbuf[8][16][40];  // per-wave P tile, padded rows

  const int h = blockIdx.y, b = blockIdx.z;
  const int tid = threadIdx.x, wave = tid >> 5, lane = tid & 31;
  const int lhalf = lane >> 4, l16 = lane & 15;

  if (tid < 32) sh_bias[tid] = rel[tid * kH + h];
  __syncthreads();

  const int bh = b * kH + h;
  const unsigned short* __restrict__ qb = qbf + (size_t)bh * kS * kDK;
  const unsigned short* __restrict__ kb = kbf + (size_t)bh * kS * kDK;
  const unsigned short* __restrict__ vb = vtbf + (size_t)bh * kDK * kS;

  const int q0 = blockIdx.x * 128 + wave * 16;

  // Q fragments (A operand), row = lane%16, 2 k-chunks of 32 over DK=64
  Frag qa[2];
  {
    const unsigned short* qrow = qb + (q0 + l16) * kDK;
#pragma unroll
    for (int kc = 0; kc < 2; ++kc) {
      qa[kc].q[0] = *(const uint4*)(qrow + kc * 32 + lhalf * 16);
      qa[kc].q[1] = *(const uint4*)(qrow + kc * 32 + lhalf * 16 + 8);
    }
  }

  f32x8 o[4];
#pragma unroll
  for (int t = 0; t < 4; ++t)
#pragma unroll
    for (int r = 0; r < 8; ++r) o[t][r] = 0.0f;

  float mrow[8], lrow[8];
#pragma unroll
  for (int r = 0; r < 8; ++r) { mrow[r] = -1e30f; lrow[r] = 0.0f; }

  const int jmax = q0 + 15;
  for (int j0 = 0; j0 <= jmax; j0 += 32) {
    // ---- scores S = Q K^T : 2 n-tiles x 2 k-chunks -> 4 WMMA
    f32x8 sc[2];
#pragma unroll
    for (int nt = 0; nt < 2; ++nt) {
#pragma unroll
      for (int r = 0; r < 8; ++r) sc[nt][r] = 0.0f;
#pragma unroll
      for (int kc = 0; kc < 2; ++kc) {
        Frag kf;  // B operand: col n = kv pos, contiguous DK values of K row
        const unsigned short* krow = kb + (j0 + nt * 16 + l16) * kDK + kc * 32 + lhalf * 16;
        kf.q[0] = *(const uint4*)(krow);
        kf.q[1] = *(const uint4*)(krow + 8);
        sc[nt] = wmma_bf16(qa[kc], kf, sc[nt]);
      }
    }

    // ---- scale + T5 bias + causal/padding mask
#pragma unroll
    for (int nt = 0; nt < 2; ++nt) {
      const int col = j0 + nt * 16 + l16;
      const int pm = mask[b * kS + col];
#pragma unroll
      for (int r = 0; r < 8; ++r) {
        const int qr = q0 + lhalf * 8 + r;
        float s = sc[nt][r] * 0.125f;   // 1/sqrt(64)
        int delta = qr - col;
        int idx = (delta < -16 ? -16 : (delta > 15 ? 15 : delta)) + 16;
        s += sh_bias[idx];
        if (col > qr || pm == 0) s = -1e30f;
        sc[nt][r] = s;
      }
    }

    // ---- online softmax (rows live in 16-lane halves of the wave)
    float mnew[8], alpha[8];
#pragma unroll
    for (int r = 0; r < 8; ++r) {
      float v = fmaxf(sc[0][r], sc[1][r]);
#pragma unroll
      for (int off = 1; off < 16; off <<= 1) v = fmaxf(v, __shfl_xor(v, off, 32));
      mnew[r]  = fmaxf(mrow[r], v);
      alpha[r] = __expf(mrow[r] - mnew[r]);
    }
#pragma unroll
    for (int nt = 0; nt < 2; ++nt)
#pragma unroll
      for (int r = 0; r < 8; ++r) sc[nt][r] = __expf(sc[nt][r] - mnew[r]);
#pragma unroll
    for (int r = 0; r < 8; ++r) {
      float v = sc[0][r] + sc[1][r];
#pragma unroll
      for (int off = 1; off < 16; off <<= 1) v += __shfl_xor(v, off, 32);
      lrow[r] = lrow[r] * alpha[r] + v;
      mrow[r] = mnew[r];
    }
#pragma unroll
    for (int t = 0; t < 4; ++t)
#pragma unroll
      for (int r = 0; r < 8; ++r) o[t][r] *= alpha[r];

    // ---- P: C-layout -> LDS -> A-layout
#pragma unroll
    for (int nt = 0; nt < 2; ++nt)
#pragma unroll
      for (int r = 0; r < 8; ++r)
        pbuf[wave][lhalf * 8 + r][nt * 16 + l16] = f2bf(sc[nt][r]);
    asm volatile("s_wait_dscnt 0" ::: "memory");
    Frag pa;
    pa.q[0] = *(const uint4*)&pbuf[wave][l16][lhalf * 16];
    pa.q[1] = *(const uint4*)&pbuf[wave][l16][lhalf * 16 + 8];

    // ---- O += P V : 4 dk n-tiles, B from V^T (contiguous 32B per lane)
#pragma unroll
    for (int t = 0; t < 4; ++t) {
      Frag vf;
      const unsigned short* vrow = vb + (t * 16 + l16) * kS + j0 + lhalf * 16;
      vf.q[0] = *(const uint4*)(vrow);
      vf.q[1] = *(const uint4*)(vrow + 8);
      o[t] = wmma_bf16(pa, vf, o[t]);
    }
  }

  // ---- normalize + store ctx (bf16, [B][S][HID])
#pragma unroll
  for (int t = 0; t < 4; ++t)
#pragma unroll
    for (int r = 0; r < 8; ++r) {
      const int srow = q0 + lhalf * 8 + r;
      float val = o[t][r] / lrow[r];
      ctx[(size_t)(b * kS + srow) * kHID + h * kDK + t * 16 + l16] = f2bf(val);
    }
}

// ---------------------------------------------------------------------------
// Output projection: out = ctx @ Wo.T (f32 output straight to d_out),
// Wo tile staged in LDS via async loads.
// ---------------------------------------------------------------------------
__global__ __launch_bounds__(256)
void proj_out_kernel(const unsigned short* __restrict__ cbf,
                     const unsigned short* __restrict__ wo,
                     float* __restrict__ out) {
  __shared__ __align__(16) unsigned short tileB[64][32];

  const int tid = threadIdx.x, wave = tid >> 5, lane = tid & 31;
  const int lhalf = lane >> 4, l16 = lane & 15;
  const int m0 = blockIdx.x * 128 + (wave & 3) * 32;
  const int nblk = blockIdx.y * 64;
  const int wn = wave >> 2;

  f32x8 acc[2][2];
#pragma unroll
  for (int mt = 0; mt < 2; ++mt)
#pragma unroll
    for (int nt = 0; nt < 2; ++nt)
#pragma unroll
      for (int r = 0; r < 8; ++r) acc[mt][nt][r] = 0.0f;

  const unsigned lds_base = (unsigned)(unsigned long long)(void*)&tileB[0][0];
  const int tn = tid >> 2;
  const int tc = tid & 3;
  const unsigned short* gsrc = wo + (size_t)(nblk + tn) * kHID + tc * 8;
  const unsigned lds_dst = lds_base + tn * 64 + tc * 16;

  for (int k0 = 0; k0 < kHID; k0 += 32) {
    asm volatile("global_load_async_to_lds_b128 %0, %1, off"
                 :: "v"(lds_dst), "v"(gsrc + k0) : "memory");
    asm volatile("s_wait_asynccnt 0" ::: "memory");
    __syncthreads();

    Frag a[2], bb[2];
#pragma unroll
    for (int mt = 0; mt < 2; ++mt) {
      const unsigned short* arow = cbf + (m0 + mt * 16 + l16) * kHID + k0;
      a[mt].q[0] = *(const uint4*)(arow + lhalf * 16);
      a[mt].q[1] = *(const uint4*)(arow + lhalf * 16 + 8);
    }
#pragma unroll
    for (int nt = 0; nt < 2; ++nt) {
      const unsigned short* brow = &tileB[wn * 32 + nt * 16 + l16][lhalf * 16];
      bb[nt].q[0] = *(const uint4*)(brow);
      bb[nt].q[1] = *(const uint4*)(brow + 8);
    }
#pragma unroll
    for (int mt = 0; mt < 2; ++mt)
#pragma unroll
      for (int nt = 0; nt < 2; ++nt)
        acc[mt][nt] = wmma_bf16(a[mt], bb[nt], acc[mt][nt]);

    __syncthreads();
  }

#pragma unroll
  for (int mt = 0; mt < 2; ++mt)
#pragma unroll
    for (int nt = 0; nt < 2; ++nt)
#pragma unroll
      for (int r = 0; r < 8; ++r) {
        int m = m0 + mt * 16 + lhalf * 8 + r;
        int n = nblk + wn * 32 + nt * 16 + l16;
        out[(size_t)m * kHID + n] = acc[mt][nt][r];
      }
}

// ---------------------------------------------------------------------------
extern "C" void kernel_launch(void* const* d_in, const int* in_sizes, int n_in,
                              void* d_out, int out_size, void* d_ws, size_t ws_size,
                              hipStream_t stream) {
  const float* x    = (const float*)d_in[0];
  const int*   mask = (const int*)  d_in[1];
  const float* Wq   = (const float*)d_in[2];
  const float* Wk   = (const float*)d_in[3];
  const float* Wv   = (const float*)d_in[4];
  const float* Wo   = (const float*)d_in[5];
  const float* rel  = (const float*)d_in[6];
  float* out = (float*)d_out;

  // workspace carve-up (bf16 buffers), 256B aligned
  char* ws = (char*)d_ws;
  size_t off = 0;
  auto take = [&](size_t bytes) {
    char* p = ws + off;
    off += (bytes + 255) & ~(size_t)255;
    return (unsigned short*)p;
  };
  unsigned short* xbf  = take((size_t)kM * kHID * 2);
  unsigned short* wqbf = take((size_t)kHID * kHID * 2);
  unsigned short* wkbf = take((size_t)kHID * kHID * 2);
  unsigned short* wvbf = take((size_t)kHID * kHID * 2);
  unsigned short* wobf = take((size_t)kHID * kHID * 2);
  unsigned short* qbf  = take((size_t)kB * kH * kS * kDK * 2);
  unsigned short* kbf  = take((size_t)kB * kH * kS * kDK * 2);
  unsigned short* vtbf = take((size_t)kB * kH * kDK * kS * 2);
  unsigned short* cbf  = take((size_t)kM * kHID * 2);

  // 1) casts
  cast_bf16_kernel<<<dim3(2048), dim3(256), 0, stream>>>(x, xbf, kM * kHID);
  cast_bf16_kernel<<<dim3(1024), dim3(256), 0, stream>>>(Wq, wqbf, kHID * kHID);
  cast_bf16_kernel<<<dim3(1024), dim3(256), 0, stream>>>(Wk, wkbf, kHID * kHID);
  cast_bf16_kernel<<<dim3(1024), dim3(256), 0, stream>>>(Wv, wvbf, kHID * kHID);
  cast_bf16_kernel<<<dim3(1024), dim3(256), 0, stream>>>(Wo, wobf, kHID * kHID);

  // 2) Q/K/V projections (z picks weight): M/128 x N/64 x 3
  proj_qkv_kernel<<<dim3(kM / 128, kHID / 64, 3), dim3(256), 0, stream>>>(
      xbf, wqbf, wkbf, wvbf, qbf, kbf, vtbf);

  // 3) attention: (S/128, H, B)
  attn_kernel<<<dim3(kS / 128, kH, kB), dim3(256), 0, stream>>>(
      qbf, kbf, vtbf, mask, rel, cbf);

  // 4) output projection
  proj_out_kernel<<<dim3(kM / 128, kHID / 64, 1), dim3(256), 0, stream>>>(
      cbf, wobf, out);
}
